// StochasticTwoLayerRGCN_attention_35648228557573
// MI455X (gfx1250) — matrix operand order repaired and checked by reading, not compiled, not run on hardware
//
#include <hip/hip_runtime.h>
#include <hip/hip_fp16.h>

typedef __attribute__((ext_vector_type(16))) _Float16 v16h;
typedef __attribute__((ext_vector_type(8)))  float    v8f;

#define DD   64
#define RREL 4
#define HATT 32

__device__ __forceinline__ void gatomicAdd(float* p, float v) {
    // lowers to global_atomic_add_f32 (no CAS loop)
    unsafeAtomicAdd(p, v);
}

// ---------------------------------------------------------------------------
// Y[nrows,64] = X[nrows,64] @ W[64,64]
// W is pre-swizzled f32->f16 into LDS in the exact WMMA B-fragment layout:
// fragment id = kblock*128 + nt*32 + lane, 16 halves (32 B) contiguous,
// so each lane fetches a fragment with two ds_load_b128.
// One wave -> 16 rows x 64 cols; K=64 = 2 k-steps of 32; 8 WMMAs per wave.
// ---------------------------------------------------------------------------
__global__ void __launch_bounds__(256)
gemm64_wmma(const float* __restrict__ X, const float* __restrict__ W,
            float* __restrict__ Y, int nrows) {
    __shared__ _Float16 swz[2 * 4 * 32 * 16];          // 8 KB, 256 fragments
    const int tid = threadIdx.x;

    // --- stage: thread t builds fragment t -------------------------------
    {
        const int kb = tid >> 7;                // k-block (0..1)
        const int nt = (tid >> 5) & 3;          // n-tile  (0..3)
        const int ln = tid & 31;                // lane the fragment feeds
        const int hf = ln >> 4, mm = ln & 15;
        const int n  = nt * 16 + mm;
        v16h frag;
#pragma unroll
        for (int i = 0; i < 16; ++i)
            frag[i] = (_Float16)W[(kb * 32 + (i >> 3) * 16 + hf * 8 + (i & 7)) * 64 + n];
        ((v16h*)swz)[tid] = frag;
    }
    __syncthreads();

    const int wave = tid >> 5, lane = tid & 31;
    const int half = lane >> 4, m = lane & 15;
    const long rowBase = (long)blockIdx.x * 128 + wave * 16;
    if (rowBase >= nrows) return;                      // wave-uniform

    // --- A fragments (16x32 f16, ISA layout) via float4 global loads -----
    const float4* xr4 = (const float4*)(X + (rowBase + m) * 64);
    const float4 p0 = xr4[half * 2 + 0];       // xrow[half*8 + 0..3]
    const float4 p1 = xr4[half * 2 + 1];       // xrow[half*8 + 4..7]
    const float4 p2 = xr4[4 + half * 2 + 0];   // xrow[16 + half*8 + 0..3]
    const float4 p3 = xr4[4 + half * 2 + 1];   // xrow[16 + half*8 + 4..7]
    const float4 p4 = xr4[8 + half * 2 + 0];   // xrow[32 + half*8 + 0..3]
    const float4 p5 = xr4[8 + half * 2 + 1];
    const float4 p6 = xr4[12 + half * 2 + 0];  // xrow[48 + half*8 + 0..3]
    const float4 p7 = xr4[12 + half * 2 + 1];

    v16h a0, a1;
    a0[0]  = (_Float16)p0.x; a0[1]  = (_Float16)p0.y; a0[2]  = (_Float16)p0.z; a0[3]  = (_Float16)p0.w;
    a0[4]  = (_Float16)p1.x; a0[5]  = (_Float16)p1.y; a0[6]  = (_Float16)p1.z; a0[7]  = (_Float16)p1.w;
    a0[8]  = (_Float16)p2.x; a0[9]  = (_Float16)p2.y; a0[10] = (_Float16)p2.z; a0[11] = (_Float16)p2.w;
    a0[12] = (_Float16)p3.x; a0[13] = (_Float16)p3.y; a0[14] = (_Float16)p3.z; a0[15] = (_Float16)p3.w;
    a1[0]  = (_Float16)p4.x; a1[1]  = (_Float16)p4.y; a1[2]  = (_Float16)p4.z; a1[3]  = (_Float16)p4.w;
    a1[4]  = (_Float16)p5.x; a1[5]  = (_Float16)p5.y; a1[6]  = (_Float16)p5.z; a1[7]  = (_Float16)p5.w;
    a1[8]  = (_Float16)p6.x; a1[9]  = (_Float16)p6.y; a1[10] = (_Float16)p6.z; a1[11] = (_Float16)p6.w;
    a1[12] = (_Float16)p7.x; a1[13] = (_Float16)p7.y; a1[14] = (_Float16)p7.z; a1[15] = (_Float16)p7.w;

    const v16h* bfrag = (const v16h*)swz;
#pragma unroll
    for (int nt = 0; nt < 4; ++nt) {
        const v16h b0 = bfrag[nt * 32 + lane];          // kblock 0
        const v16h b1 = bfrag[128 + nt * 32 + lane];    // kblock 1
        v8f c = {};
        c = __builtin_amdgcn_wmma_f32_16x16x32_f16(false, a0, false, b0,
                                                   (short)0, c, false, false);
        c = __builtin_amdgcn_wmma_f32_16x16x32_f16(false, a1, false, b1,
                                                   (short)0, c, false, false);
        // C/D layout: VGPR v, lanes 0-15 -> M=v ; lanes 16-31 -> M=v+8
        const int n = nt * 16 + m;
#pragma unroll
        for (int vv = 0; vv < 8; ++vv)
            Y[(rowBase + half * 8 + vv) * 64 + n] = c[vv];
    }
}

// ---------------------------------------------------------------------------
__global__ void degree_kernel(const int* __restrict__ dst, float* __restrict__ deg,
                              int E, int N) {
    long t = (long)blockIdx.x * blockDim.x + threadIdx.x;
    if (t >= (long)RREL * E) return;
    int r = (int)(t / E);
    gatomicAdd(deg + (long)r * N + dst[t], 1.0f);
}

// 16 threads per edge; float4 gather from xw (L2-resident) + 4 hw f32 atomics.
__global__ void scatter_kernel(const float4* __restrict__ xw,
                               const int* __restrict__ src,
                               const int* __restrict__ dst,
                               float* __restrict__ agg, int E) {
    long t = (long)blockIdx.x * blockDim.x + threadIdx.x;
    if (t >= (long)E * 16) return;
    const int e = (int)(t >> 4), p = (int)(t & 15);
    const int s = src[e], d = dst[e];
    const float4 v = xw[(long)s * 16 + p];
    float* out = agg + (long)d * 64 + p * 4;
    gatomicAdd(out + 0, v.x);
    gatomicAdd(out + 1, v.y);
    gatomicAdd(out + 2, v.z);
    gatomicAdd(out + 3, v.w);
}

// z[r][n][:] = relu(agg[r][n][:] / max(deg,1) + b[r][:])   (in place, no race)
__global__ void norm_bias_relu(float4* __restrict__ z, const float* __restrict__ deg,
                               const float4* __restrict__ b, int N) {
    long t = (long)blockIdx.x * blockDim.x + threadIdx.x;
    if (t >= (long)RREL * N * 16) return;
    const int q = (int)(t & 15);
    const long rn = t >> 4;
    const int r = (int)(rn / N);
    float dg = deg[rn];
    if (dg < 1.f) dg = 1.f;
    const float inv = 1.f / dg;
    float4 v = z[t];
    const float4 bb = b[r * 16 + q];
    v.x = fmaxf(v.x * inv + bb.x, 0.f);
    v.y = fmaxf(v.y * inv + bb.y, 0.f);
    v.z = fmaxf(v.z * inv + bb.z, 0.f);
    v.w = fmaxf(v.w * inv + bb.w, 0.f);
    z[t] = v;
}

// w[n,r] = tanh(z[r][n][:] @ Wp + bp) @ v ; accumulate sum over n into wsum[r]
__global__ void att_score(const float* __restrict__ z, const float* __restrict__ Wp,
                          const float* __restrict__ bp, const float* __restrict__ vv,
                          float* __restrict__ wsum, int N) {
    __shared__ float sacc[RREL];
    if (threadIdx.x < RREL) sacc[threadIdx.x] = 0.f;
    __syncthreads();
    long t = (long)blockIdx.x * blockDim.x + threadIdx.x;
    if (t < (long)N * RREL) {
        const int r = (int)(t & 3);
        const long n = t >> 2;
        const float* zr = z + ((long)r * N + n) * 64;
        float zrow[64];
#pragma unroll
        for (int d = 0; d < 64; ++d) zrow[d] = zr[d];
        float acc = 0.f;
#pragma unroll 2
        for (int j = 0; j < HATT; ++j) {
            float hh = bp[j];
#pragma unroll
            for (int d = 0; d < 64; ++d) hh = fmaf(zrow[d], Wp[d * HATT + j], hh);
            acc += tanhf(hh) * vv[j];
        }
        atomicAdd(&sacc[r], acc);          // ds_add_f32
    }
    __syncthreads();
    if (threadIdx.x < RREL) gatomicAdd(&wsum[threadIdx.x], sacc[threadIdx.x]);
}

__global__ void beta_kernel(const float* __restrict__ wsum, float* __restrict__ beta,
                            float invN) {
    if (threadIdx.x == 0 && blockIdx.x == 0) {
        float w[RREL], mx = -1e30f;
        for (int r = 0; r < RREL; ++r) { w[r] = wsum[r] * invN; mx = fmaxf(mx, w[r]); }
        float s = 0.f;
        for (int r = 0; r < RREL; ++r) { w[r] = __expf(w[r] - mx); s += w[r]; }
        const float is = 1.f / s;
        for (int r = 0; r < RREL; ++r) beta[r] = w[r] * is;
    }
}

__global__ void weighted_sum(const float4* __restrict__ z, const float* __restrict__ beta,
                             float4* __restrict__ out, int N) {
    long t = (long)blockIdx.x * blockDim.x + threadIdx.x;
    if (t >= (long)N * 16) return;
    const float b0 = beta[0], b1 = beta[1], b2 = beta[2], b3 = beta[3];
    const long stride = (long)N * 16;
    const float4 v0 = z[t], v1 = z[stride + t], v2 = z[2 * stride + t], v3 = z[3 * stride + t];
    float4 acc;
    acc.x = b0 * v0.x + b1 * v1.x + b2 * v2.x + b3 * v3.x;
    acc.y = b0 * v0.y + b1 * v1.y + b2 * v2.y + b3 * v3.y;
    acc.z = b0 * v0.z + b1 * v1.z + b2 * v2.z + b3 * v3.z;
    acc.w = b0 * v0.w + b1 * v1.w + b2 * v2.w + b3 * v3.w;
    out[t] = acc;
}

// ---------------------------------------------------------------------------
extern "C" void kernel_launch(void* const* d_in, const int* in_sizes, int n_in,
                              void* d_out, int out_size, void* d_ws, size_t ws_size,
                              hipStream_t stream) {
    const float* x   = (const float*)d_in[0];
    const int*   src = (const int*)d_in[1];
    const int*   dst = (const int*)d_in[2];
    const float* W1  = (const float*)d_in[3];
    const float* b1  = (const float*)d_in[4];
    const float* a1W = (const float*)d_in[5];
    const float* a1b = (const float*)d_in[6];
    const float* a1v = (const float*)d_in[7];
    const float* W2  = (const float*)d_in[8];
    const float* b2  = (const float*)d_in[9];
    const float* a2W = (const float*)d_in[10];
    const float* a2b = (const float*)d_in[11];
    const float* a2v = (const float*)d_in[12];

    const int N = in_sizes[0] / DD;
    const int E = in_sizes[1] / RREL;
    float* out = (float*)d_out;

    // workspace layout
    char*  ws       = (char*)d_ws;
    float* agg      = (float*)ws;                         // [R,N,64] (reused as z)
    size_t aggBytes = (size_t)RREL * N * DD * sizeof(float);
    float* xw       = (float*)(ws + aggBytes);            // [N,64]
    size_t xwBytes  = (size_t)N * DD * sizeof(float);
    float* h        = (float*)(ws + aggBytes + xwBytes);  // [N,64]
    float* deg      = (float*)(ws + aggBytes + 2 * xwBytes);  // [R,N]
    size_t degBytes = (size_t)RREL * N * sizeof(float);
    float* wsum     = (float*)(ws + aggBytes + 2 * xwBytes + degBytes);  // 4
    float* beta     = wsum + 8;                                          // 4

    const int TB = 256;

    // degrees: identical for both layers, compute once
    hipMemsetAsync(deg, 0, degBytes, stream);
    {
        const long tot = (long)RREL * E;
        degree_kernel<<<(int)((tot + TB - 1) / TB), TB, 0, stream>>>(dst, deg, E, N);
    }

    const int  gemmBlocks = (N + 127) / 128;
    const long scatT      = (long)E * 16;
    const int  scatBlocks = (int)((scatT + TB - 1) / TB);

    for (int layer = 0; layer < 2; ++layer) {
        const float* X  = (layer == 0) ? x   : h;
        const float* W  = (layer == 0) ? W1  : W2;
        const float* bb = (layer == 0) ? b1  : b2;
        const float* aW = (layer == 0) ? a1W : a2W;
        const float* ab = (layer == 0) ? a1b : a2b;
        const float* av = (layer == 0) ? a1v : a2v;
        float* dstbuf   = (layer == 0) ? h   : out;

        hipMemsetAsync(agg, 0, aggBytes, stream);
        for (int r = 0; r < RREL; ++r) {
            gemm64_wmma<<<gemmBlocks, TB, 0, stream>>>(X, W + (size_t)r * DD * DD, xw, N);
            scatter_kernel<<<scatBlocks, TB, 0, stream>>>((const float4*)xw,
                                                          src + (size_t)r * E,
                                                          dst + (size_t)r * E,
                                                          agg + (size_t)r * N * DD, E);
        }
        {
            const long tot = (long)RREL * N * 16;
            norm_bias_relu<<<(int)((tot + TB - 1) / TB), TB, 0, stream>>>(
                (float4*)agg, deg, (const float4*)bb, N);
        }
        hipMemsetAsync(wsum, 0, RREL * sizeof(float), stream);
        {
            const long tot = (long)N * RREL;
            att_score<<<(int)((tot + TB - 1) / TB), TB, 0, stream>>>(agg, aW, ab, av, wsum, N);
        }
        beta_kernel<<<1, 32, 0, stream>>>(wsum, beta, 1.0f / (float)N);
        {
            const long tot = (long)N * 16;
            weighted_sum<<<(int)((tot + TB - 1) / TB), TB, 0, stream>>>(
                (const float4*)agg, beta, (float4*)dstbuf, N);
        }
    }
}
